// TemporalAttentionModel_12670153523501
// MI455X (gfx1250) — matrix operand order repaired and checked by reading, not compiled, not run on hardware
//
#include <hip/hip_runtime.h>

typedef __attribute__((ext_vector_type(16))) _Float16 v16h;
typedef __attribute__((ext_vector_type(8)))  _Float16 v8h;
typedef __attribute__((ext_vector_type(4)))  _Float16 v4h;
typedef __attribute__((ext_vector_type(8)))  float    v8f;

#define BV 16
#define TV 64
#define NVX 256
#define DV 64
#define D2 128

// ---- LDS layout (units: halfs). Total 48640 halfs = 97280 bytes. ----
// Region A (0 .. 34816): sH [64][136] + sW12/13/14 [64][136] each.
//   Overlay 1: sP = 8 heads x [64][68] (exactly 34816 halfs).
//   Overlay 2: sM (MLP intermediate) [64][72].
// Region B (34816 .. 48640): sQ/sK/sV [64][72] each.
//   Overlay: sO = sQ region; sW15 = sK region; sW16 = sV region.
#define SH_S    136
#define SW_OFF  8704
#define SW_MAT  8704
#define SP_S    68
#define SP_HEAD 4352
#define SQ_OFF  34816
#define QKV_S   72
#define SK_OFF  39424
#define SV_OFF  44032
#define LDS_HALFS 48640

__device__ __forceinline__ v8f vzero8() {
    v8f v;
#pragma unroll
    for (int i = 0; i < 8; ++i) v[i] = 0.0f;
    return v;
}

// A-matrix fragment, 16x32 f16: lane m = lane&15, group g = lane>>4;
// element e -> K = kb + 16*g + e
__device__ __forceinline__ v16h ldA(const _Float16* base, int stride, int mb, int kb, int lane) {
    const int m = mb + (lane & 15);
    const int g = (lane >> 4) & 1;
    v16h a;
#pragma unroll
    for (int e = 0; e < 16; ++e) {
        const int k = kb + 16 * g + e;
        a[e] = base[m * stride + k];
    }
    return a;
}

// B-matrix fragment, 32x16 f16: lane n = lane&15, group g; element i -> K = kb + 16*g + i.
// base is row-major [N][K] (i.e., W[out][in], since B = W^T with K = in-dim).
__device__ __forceinline__ v16h ldB(const _Float16* base, int stride, int nb, int kb, int lane) {
    const int nn = nb + (lane & 15);
    const int g  = (lane >> 4) & 1;
    v16h b;
#pragma unroll
    for (int i = 0; i < 16; ++i) b[i] = base[nn * stride + kb + 16 * g + i];
    return b;
}

__device__ __forceinline__ float gelu_erf(float x) {
    return 0.5f * x * (1.0f + erff(x * 0.70710678118654752f));
}

__global__ __launch_bounds__(256)
void tam_fused_kernel(const float* __restrict__ X,   const float* __restrict__ STE,
                      const float* __restrict__ W12, const float* __restrict__ b12,
                      const float* __restrict__ W13, const float* __restrict__ b13,
                      const float* __restrict__ W14, const float* __restrict__ b14,
                      const float* __restrict__ W15, const float* __restrict__ b15,
                      const float* __restrict__ W16, const float* __restrict__ b16,
                      float* __restrict__ Y)
{
    extern __shared__ _Float16 lds[];
    const int tid  = threadIdx.x;
    const int lane = tid & 31;
    const int wave = tid >> 5;
    const int bid  = blockIdx.x;
    const int b    = bid / NVX;
    const int n    = bid % NVX;

    _Float16* sH = lds;             // [64][136]  (cols 0-63 = X, 64-127 = STE)
    _Float16* sW = lds + SW_OFF;    // 3 x [64][136]
    _Float16* sP = lds;             // overlay: 8 x [64][68]
    _Float16* sQ = lds + SQ_OFF;    // [64][72]
    _Float16* sK = lds + SK_OFF;    // [64][72]
    _Float16* sV = lds + SV_OFF;    // [64][72]
    _Float16* sO = lds + SQ_OFF;    // overlay of sQ
    _Float16* sW5 = lds + SK_OFF;   // overlay of sK
    _Float16* sW6 = lds + SV_OFF;   // overlay of sV
    _Float16* sM = lds;             // overlay of region A: [64][72]

    // Warm L2 for the phase-3 weights (global_prefetch_b8); they are first
    // consumed 4 barriers from now, so this hides their latency completely.
    if (tid < 128) {
        __builtin_prefetch(W15 + tid * 32, 0, 0);   // 128 x 128B = 16 KB
        __builtin_prefetch(W16 + tid * 32, 0, 0);
    }

    // ---------------- Phase 0: stage H and QKV weights (f32 -> f16) ----------------
    {
        const size_t rstride = (size_t)NVX * DV;                   // t-stride in floats
        const float* Xp = X   + ((size_t)b * TV * NVX + n) * DV;
        const float* Sp = STE + ((size_t)b * TV * NVX + n) * DV;
        for (int i = tid; i < TV * 16; i += 256) {                 // 16 float4 per row
            const int t = i >> 4, c4 = i & 15;
            const float4 xv = *(const float4*)(Xp + (size_t)t * rstride + c4 * 4);
            const float4 sv = *(const float4*)(Sp + (size_t)t * rstride + c4 * 4);
            v4h px = { (_Float16)xv.x, (_Float16)xv.y, (_Float16)xv.z, (_Float16)xv.w };
            v4h ps = { (_Float16)sv.x, (_Float16)sv.y, (_Float16)sv.z, (_Float16)sv.w };
            *(v4h*)(sH + t * SH_S + c4 * 4)      = px;
            *(v4h*)(sH + t * SH_S + 64 + c4 * 4) = ps;
        }
#pragma unroll
        for (int m = 0; m < 3; ++m) {
            const float* Wp = (m == 0) ? W12 : (m == 1) ? W13 : W14;
            _Float16* dst = sW + m * SW_MAT;
            for (int i = tid; i < DV * 32; i += 256) {             // 32 float4 per row of 128
                const int o = i >> 5, c4 = i & 31;
                const float4 wv = *(const float4*)(Wp + o * D2 + c4 * 4);
                v4h pw = { (_Float16)wv.x, (_Float16)wv.y, (_Float16)wv.z, (_Float16)wv.w };
                *(v4h*)(dst + o * SH_S + c4 * 4) = pw;
            }
        }
    }
    __syncthreads();

    // ---------------- Phase 1: Q/K/V = gelu(H @ W^T + b) ----------------
    // 48 output tiles (3 mats x 4 Mtiles x 4 Ntiles), round-robin over 8 waves.
    for (int tile = wave; tile < 48; tile += 8) {
        const int mat = tile / 16;
        const int rem = tile % 16;
        const int mt = rem >> 2, nt = rem & 3;
        const _Float16* Wm = sW + mat * SW_MAT;
        v8f acc = vzero8();
#pragma unroll
        for (int kb = 0; kb < 4; ++kb) {
            v16h a = ldA(sH, SH_S, mt * 16, kb * 32, lane);
            v16h bm = ldB(Wm, SH_S, nt * 16, kb * 32, lane);
            acc = __builtin_amdgcn_wmma_f32_16x16x32_f16(false, a, false, bm,
                                                         (short)0, acc, false, false);
        }
        const float* bias = (mat == 0) ? b12 : (mat == 1) ? b13 : b14;
        const int col = nt * 16 + (lane & 15);
        const float bv = bias[col];
        _Float16* dst = (mat == 0) ? sQ : (mat == 1) ? sK : sV;
#pragma unroll
        for (int r = 0; r < 8; ++r) {
            const int row = mt * 16 + r + ((lane & 16) ? 8 : 0);
            dst[row * QKV_S + col] = (_Float16)gelu_erf(acc[r] + bv);
        }
    }
    __syncthreads();

    // ---------------- Phase 2a: per-head causal attention, P = softmax(QK^T/sqrt(d)) ----
    const int h  = wave;          // one wave per head
    const int hc = h * 8;
    _Float16* sPh = sP + h * SP_HEAD;
    const int g = (lane >> 4) & 1;
    for (int mt = 0; mt < 4; ++mt) {
        v8f S[4];
#pragma unroll
        for (int nt = 0; nt < 4; ++nt) {
            v16h a, bm;
#pragma unroll
            for (int e = 0; e < 16; ++e) { a[e] = (_Float16)0; bm[e] = (_Float16)0; }
            if (g == 0) {
                const int m = mt * 16 + (lane & 15);
                const int s = nt * 16 + (lane & 15);
                // rows are 16B-aligned at the head offset: single b128 DS load each
                const v8h qv = *(const v8h*)(sQ + m * QKV_S + hc);
                const v8h kv = *(const v8h*)(sK + s * QKV_S + hc);
#pragma unroll
                for (int e = 0; e < 8; ++e) {      // K = e (head dim, zero-padded to 32)
                    a[e]  = qv[e];
                    bm[e] = kv[e];
                }
            }
            S[nt] = __builtin_amdgcn_wmma_f32_16x16x32_f16(false, a, false, bm,
                                                           (short)0, vzero8(), false, false);
        }
        // scale + causal mask + row softmax (rows live across lanes 0-15 / 16-31 halves)
        const int scol = (lane & 15);
#pragma unroll
        for (int r = 0; r < 8; ++r) {
            const int qrow = mt * 16 + r + ((lane & 16) ? 8 : 0);
            float mx = -3.0e38f;
#pragma unroll
            for (int nt = 0; nt < 4; ++nt) {
                const int s = nt * 16 + scol;
                float x = S[nt][r] * 0.35355339059327373f;
                x = (s <= qrow) ? x : -3.0e38f;
                S[nt][r] = x;
                mx = fmaxf(mx, x);
            }
#pragma unroll
            for (int off = 8; off >= 1; off >>= 1) mx = fmaxf(mx, __shfl_xor(mx, off, 32));
            float sum = 0.0f;
#pragma unroll
            for (int nt = 0; nt < 4; ++nt) {
                const float p = __expf(S[nt][r] - mx);
                S[nt][r] = p;
                sum += p;
            }
#pragma unroll
            for (int off = 8; off >= 1; off >>= 1) sum += __shfl_xor(sum, off, 32);
            const float inv = 1.0f / sum;
#pragma unroll
            for (int nt = 0; nt < 4; ++nt) {
                const int row = mt * 16 + r + ((lane & 16) ? 8 : 0);
                sPh[row * SP_S + nt * 16 + scol] = (_Float16)(S[nt][r] * inv);
            }
        }
    }
    __syncthreads();

    // ---------------- Phase 2b: O_h = P_h @ V_h  (writes sO = sQ region) ----------------
    for (int mt = 0; mt < 4; ++mt) {
        v8f acc = vzero8();
#pragma unroll
        for (int kb = 0; kb < 2; ++kb) {
            v16h a = ldA(sPh, SP_S, mt * 16, kb * 32, lane);
            v16h bm;
#pragma unroll
            for (int i = 0; i < 16; ++i) bm[i] = (_Float16)0;
            const int nn = lane & 15;
            if (nn < 8) {
#pragma unroll
                for (int i = 0; i < 16; ++i) {
                    const int s = kb * 32 + 16 * g + i;            // time axis K
                    bm[i] = sV[s * QKV_S + hc + nn];
                }
            }
            acc = __builtin_amdgcn_wmma_f32_16x16x32_f16(false, a, false, bm,
                                                         (short)0, acc, false, false);
        }
        if ((lane & 15) < 8) {
#pragma unroll
            for (int r = 0; r < 8; ++r) {
                const int row = mt * 16 + r + ((lane & 16) ? 8 : 0);
                sO[row * QKV_S + hc + (lane & 15)] = (_Float16)acc[r];
            }
        }
    }
    __syncthreads();

    // ---------------- Phase 3: stage W15/W16 (overlay sK/sV) ----------------
#pragma unroll
    for (int m = 0; m < 2; ++m) {
        const float* Wp = (m == 0) ? W15 : W16;
        _Float16* dst = (m == 0) ? sW5 : sW6;
        for (int i = tid; i < DV * 16; i += 256) {                 // 16 float4 per row of 64
            const int o = i >> 4, c4 = i & 15;
            const float4 wv = *(const float4*)(Wp + o * DV + c4 * 4);
            v4h pw = { (_Float16)wv.x, (_Float16)wv.y, (_Float16)wv.z, (_Float16)wv.w };
            *(v4h*)(dst + o * QKV_S + c4 * 4) = pw;
        }
    }
    __syncthreads();

    // MLP layer 1: sM = gelu(sO @ W15^T + b15)   (sM overlays region A)
    for (int tile = wave; tile < 16; tile += 8) {
        const int mt = tile >> 2, nt = tile & 3;
        v8f acc = vzero8();
#pragma unroll
        for (int kb = 0; kb < 2; ++kb) {
            v16h a  = ldA(sO, QKV_S, mt * 16, kb * 32, lane);
            v16h bm = ldB(sW5, QKV_S, nt * 16, kb * 32, lane);
            acc = __builtin_amdgcn_wmma_f32_16x16x32_f16(false, a, false, bm,
                                                         (short)0, acc, false, false);
        }
        const int col = nt * 16 + (lane & 15);
        const float bv = b15[col];
#pragma unroll
        for (int r = 0; r < 8; ++r) {
            const int row = mt * 16 + r + ((lane & 16) ? 8 : 0);
            sM[row * QKV_S + col] = (_Float16)gelu_erf(acc[r] + bv);
        }
    }
    __syncthreads();

    // MLP layer 2: Y = sM @ W16^T + b16  -> global (fp32)
    for (int tile = wave; tile < 16; tile += 8) {
        const int mt = tile >> 2, nt = tile & 3;
        v8f acc = vzero8();
#pragma unroll
        for (int kb = 0; kb < 2; ++kb) {
            v16h a  = ldA(sM, QKV_S, mt * 16, kb * 32, lane);
            v16h bm = ldB(sW6, QKV_S, nt * 16, kb * 32, lane);
            acc = __builtin_amdgcn_wmma_f32_16x16x32_f16(false, a, false, bm,
                                                         (short)0, acc, false, false);
        }
        const int col = nt * 16 + (lane & 15);
        const float bv = b16[col];
#pragma unroll
        for (int r = 0; r < 8; ++r) {
            const int row = mt * 16 + r + ((lane & 16) ? 8 : 0);
            Y[(((size_t)b * TV + row) * NVX + n) * DV + col] = acc[r] + bv;
        }
    }
}

extern "C" void kernel_launch(void* const* d_in, const int* in_sizes, int n_in,
                              void* d_out, int out_size, void* d_ws, size_t ws_size,
                              hipStream_t stream) {
    (void)in_sizes; (void)n_in; (void)out_size; (void)d_ws; (void)ws_size;
    const float* X   = (const float*)d_in[0];
    const float* STE = (const float*)d_in[1];
    const float* W12 = (const float*)d_in[2];
    const float* b12 = (const float*)d_in[3];
    const float* W13 = (const float*)d_in[4];
    const float* b13 = (const float*)d_in[5];
    const float* W14 = (const float*)d_in[6];
    const float* b14 = (const float*)d_in[7];
    const float* W15 = (const float*)d_in[8];
    const float* b15 = (const float*)d_in[9];
    const float* W16 = (const float*)d_in[10];
    const float* b16 = (const float*)d_in[11];

    const dim3 grid(BV * NVX);          // 4096 workgroups: one per (batch, vertex)
    const dim3 block(256);              // 8 wave32s
    const size_t shmem = LDS_HALFS * sizeof(_Float16);  // 97280 B (< 320 KB/WGP, 3 blocks/WGP)
    tam_fused_kernel<<<grid, block, shmem, stream>>>(X, STE, W12, b12, W13, b13,
                                                     W14, b14, W15, b15, W16, b16,
                                                     (float*)d_out);
}
